// TemporalAttention_29420525978033
// MI455X (gfx1250) — matrix operand order repaired and checked by reading, not compiled, not run on hardware
//
#include <hip/hip_runtime.h>
#include <hip/hip_bf16.h>

typedef _Float16 h4   __attribute__((ext_vector_type(4)));
typedef _Float16 h8   __attribute__((ext_vector_type(8)));
typedef _Float16 v16h __attribute__((ext_vector_type(16)));
typedef float    v8f  __attribute__((ext_vector_type(8)));

#define WMMA_F16(A, B, C) \
  __builtin_amdgcn_wmma_f32_16x16x32_f16(false, (A), false, (B), (short)0, (C), false, false)

// Problem constants
#define SQ     64     // sequence length
#define DM     256    // model dim
#define NH     8      // heads
#define HD     32     // head dim
#define PITCH  264    // LDS row pitch (halves) for X/Q/K/O: 528B rows, 16B aligned, bank-friendly
#define VPITCH 72     // per-head V^T row pitch (halves)
#define PPITCH 68     // per-wave P row pitch (halves): 34-dword stride -> conflict-free rows

// f32 x-tile staging region (64KB) overlaps Vt+Pb (dead during phase 0).
#define STAGE_HALVES (3 * SQ * PITCH)            // 50688 halves
#define STAGE_BYTES  (STAGE_HALVES * 2)          // 101376 B

// ---- A-fragment (f16 16x32, contiguous k order, same as B): lane holds row M=lane%16.
// halves 0..7  = K in [col0 + hi*16, +8)        (vgpr0..3)
// halves 8..15 = K in [col0 + hi*16 + 8, +8)    (vgpr4..7)
__device__ __forceinline__ v16h lds_a_frag(const _Float16* row_ptr, int hi) {
  const _Float16* p = row_ptr + hi * 16;
  h8 lo = *(const h8*)p;
  h8 hh = *(const h8*)(p + 8);
  v16h r;
#pragma unroll
  for (int t = 0; t < 8; ++t) { r[t] = lo[t]; r[8 + t] = hh[t]; }
  return r;
}

// ---- B-fragment (f16 32x16): lane holds column N=lane%16; 16 consecutive K
// values starting at col0 + hi*16. Column of B == a row of the LDS array.
__device__ __forceinline__ v16h lds_b_frag(const _Float16* row_ptr, int hi) {
  const _Float16* p = row_ptr + hi * 16;
  h8 lo = *(const h8*)p;
  h8 hh = *(const h8*)(p + 8);
  v16h r;
#pragma unroll
  for (int t = 0; t < 8; ++t) { r[t] = lo[t]; r[8 + t] = hh[t]; }
  return r;
}

__device__ __forceinline__ v16h glb_b_frag(const _Float16* p, int hi) {
  return *(const v16h*)(p + hi * 16);  // 32B-aligned global b128 pair
}

// ---------------- weight fp32 -> fp16 pre-pass ----------------
__global__ __launch_bounds__(256) void wconvert_kernel(
    const float* __restrict__ wq, const float* __restrict__ wp,
    _Float16* __restrict__ oq, _Float16* __restrict__ op) {
  int i = blockIdx.x * 256 + threadIdx.x;
  if (i < 3 * DM * DM) oq[i] = (_Float16)wq[i];
  if (i < DM * DM)     op[i] = (_Float16)wp[i];
}

// ---------------- fused temporal attention ----------------
__global__ __launch_bounds__(256) void temporal_attn_kernel(
    const float* __restrict__ x,
    const float* __restrict__ b_qkv,
    const float* __restrict__ b_proj,
    const _Float16* __restrict__ Wq,   // [768,256] f16 row-major
    const _Float16* __restrict__ Wp,   // [256,256] f16 row-major
    float* __restrict__ out) {
  extern __shared__ _Float16 sm[];
  _Float16* Xh = sm;                    // [64][264] x tile (f16); reused as attention output O
  _Float16* Qh = sm + SQ * PITCH;       // [64][264] Q (pre-scaled)
  _Float16* Kh = sm + 2 * SQ * PITCH;   // [64][264] K
  _Float16* Vt = sm + 3 * SQ * PITCH;   // [8][32][72] per-head V^T
  _Float16* Pb = Vt + NH * HD * VPITCH; // [8][64][68] per-wave softmax probs

  const int tid  = threadIdx.x;
  const int lane = tid & 31;
  const int wid  = tid >> 5;            // 8 waves
  const int ln   = lane & 15;
  const int hi   = lane >> 4;           // which 16-lane half
  const size_t base = (size_t)blockIdx.x * (SQ * DM);

  // Warm L2/WGP$ with the weight rows this wave will consume (global_prefetch_b8),
  // overlapping with the async x-tile load below.
#pragma unroll
  for (int eb = 0; eb < 6; ++eb)
    __builtin_prefetch(Wq + (size_t)(16 * (wid + 8 * eb) + ln) * DM, 0, 3);
#pragma unroll
  for (int eb = 0; eb < 2; ++eb)
    __builtin_prefetch(Wp + (size_t)(16 * (wid + 8 * eb) + ln) * DM, 0, 3);

  // ---- Phase 0: async-copy x tile (f32) global -> LDS staging, no VGPR round-trip ----
  {
    const float* xb = x + base;               // uniform 64-bit base (SGPR pair)
    unsigned goff = (unsigned)tid * 16u;
#pragma unroll
    for (int k = 0; k < 16; ++k) {
      unsigned go = goff + (unsigned)k * 4096u;
      unsigned lo = (unsigned)STAGE_BYTES + go;
      asm volatile("global_load_async_to_lds_b128 %0, %1, %2"
                   :: "v"(lo), "v"(go), "s"(xb) : "memory");
    }
    asm volatile("s_wait_asynccnt 0x0" ::: "memory");
    // convert staged f32 -> f16 X tile (LDS -> LDS)
    const float4* stg = (const float4*)(sm + STAGE_HALVES);
#pragma unroll
    for (int k = 0; k < 16; ++k) {
      int idx = tid + k * 256;                // float4 index over 64*256 floats
      float4 v = stg[idx];
      int i4 = idx * 4, row = i4 >> 8, col = i4 & 255;
      h4 hv = {(_Float16)v.x, (_Float16)v.y, (_Float16)v.z, (_Float16)v.w};
      *(h4*)(Xh + row * PITCH + col) = hv;
    }
  }
  __syncthreads();

  // ---- Phase 1: QKV GEMM  [64,256] x [256,768]^T + bias ----
  // A fragments held resident per row-tile; B fragments batch-loaded from
  // global so the 8-WMMA chain runs with staggered loadcnt waits.
  const float qscale = 0.17677669529663687f;  // 1/sqrt(32), folded into Q
  for (int mt = 0; mt < 4; ++mt) {
    const _Float16* arow = Xh + (16 * mt + ln) * PITCH;
    v16h Af[8];
#pragma unroll
    for (int kk = 0; kk < 8; ++kk) Af[kk] = lds_a_frag(arow + 32 * kk, hi);
    for (int eb = 0; eb < 6; ++eb) {
      int et = wid + 8 * eb;                       // e-tile 0..47
      const _Float16* brow = Wq + (size_t)(16 * et + ln) * DM;
      v16h Bf[8];
#pragma unroll
      for (int kk = 0; kk < 8; ++kk) Bf[kk] = glb_b_frag(brow + 32 * kk, hi);
      v8f acc = {};
#pragma unroll
      for (int kk = 0; kk < 8; ++kk) acc = WMMA_F16(Af[kk], Bf[kk], acc);
      int col = 16 * et + ln;                      // output column e in [0,768)
      float bias = b_qkv[col];
#pragma unroll
      for (int r = 0; r < 8; ++r) {
        int srow = 16 * mt + r + 8 * hi;           // token index s
        float v = acc[r] + bias;
        if (et < 16) {                             // Q block, pre-scale
          Qh[srow * PITCH + col] = (_Float16)(v * qscale);
        } else if (et < 32) {                      // K block
          Kh[srow * PITCH + (col - 256)] = (_Float16)v;
        } else {                                   // V block -> per-head transpose
          int ev = col - 512, hh = ev >> 5, el = ev & 31;
          Vt[hh * (HD * VPITCH) + el * VPITCH + srow] = (_Float16)v;
        }
      }
    }
  }
  __syncthreads();

  // ---- Phase 2: attention, one head per wave ----
  const int h = wid;
  _Float16* P = Pb + wid * (SQ * PPITCH);
  {
    // K B-fragments are invariant across q-tiles: load once (32 VGPRs).
    v16h Bk[4];
#pragma unroll
    for (int j = 0; j < 4; ++j)
      Bk[j] = lds_b_frag(Kh + (16 * j + ln) * PITCH + 32 * h, hi);
    for (int i = 0; i < 4; ++i) {                  // q row-tile
      v16h Aq = lds_a_frag(Qh + (16 * i + ln) * PITCH + 32 * h, hi);
      v8f S[4];
#pragma unroll
      for (int j = 0; j < 4; ++j) {
        v8f z = {};
        S[j] = WMMA_F16(Aq, Bk[j], z);
      }
      // row softmax: row lives across the 16 lanes of a half (xor 1,2,4,8)
#pragma unroll
      for (int r = 0; r < 8; ++r) {
        float m = fmaxf(fmaxf(S[0][r], S[1][r]), fmaxf(S[2][r], S[3][r]));
        m = fmaxf(m, __shfl_xor(m, 1, 32));
        m = fmaxf(m, __shfl_xor(m, 2, 32));
        m = fmaxf(m, __shfl_xor(m, 4, 32));
        m = fmaxf(m, __shfl_xor(m, 8, 32));
        float p0 = __expf(S[0][r] - m), p1 = __expf(S[1][r] - m);
        float p2 = __expf(S[2][r] - m), p3 = __expf(S[3][r] - m);
        float s = p0 + p1 + p2 + p3;
        s += __shfl_xor(s, 1, 32);
        s += __shfl_xor(s, 2, 32);
        s += __shfl_xor(s, 4, 32);
        s += __shfl_xor(s, 8, 32);
        float inv = 1.0f / s;
        int q = 16 * i + r + 8 * hi;
        _Float16* pr = P + q * PPITCH + ln;
        pr[0]  = (_Float16)(p0 * inv);
        pr[16] = (_Float16)(p1 * inv);
        pr[32] = (_Float16)(p2 * inv);
        pr[48] = (_Float16)(p3 * inv);
      }
    }
  }
  // PV: [64,64] x [64,32] per head (per-wave LDS in-order => P visible)
  {
    const _Float16* VtH = Vt + h * (HD * VPITCH);
    // V B-fragments invariant across q-tiles: load once (32 VGPRs).
    v16h Bv[2][2];
#pragma unroll
    for (int jo = 0; jo < 2; ++jo)
#pragma unroll
      for (int kk = 0; kk < 2; ++kk)
        Bv[jo][kk] = lds_b_frag(VtH + (16 * jo + ln) * VPITCH + 32 * kk, hi);
    for (int i = 0; i < 4; ++i) {
      const _Float16* prow = P + (16 * i + ln) * PPITCH;
      v16h Ap[2];
#pragma unroll
      for (int kk = 0; kk < 2; ++kk) {
        const _Float16* ap = prow + 32 * kk + hi * 16;
#pragma unroll
        for (int t = 0; t < 8; ++t) { Ap[kk][t] = ap[t]; Ap[kk][8 + t] = ap[8 + t]; }
      }
#pragma unroll
      for (int jo = 0; jo < 2; ++jo) {
        v8f acc = {};
#pragma unroll
        for (int kk = 0; kk < 2; ++kk) acc = WMMA_F16(Ap[kk], Bv[jo][kk], acc);
#pragma unroll
        for (int r = 0; r < 8; ++r) {
          int q = 16 * i + r + 8 * hi;
          Xh[q * PITCH + 32 * h + 16 * jo + ln] = (_Float16)acc[r];  // O reuses Xh
        }
      }
    }
  }
  __syncthreads();

  // ---- Phase 3: output projection [64,256] x [256,256]^T + bias ----
  for (int mt = 0; mt < 4; ++mt) {
    const _Float16* arow = Xh + (16 * mt + ln) * PITCH;
    v16h Af[8];
#pragma unroll
    for (int kk = 0; kk < 8; ++kk) Af[kk] = lds_a_frag(arow + 32 * kk, hi);
#pragma unroll
    for (int eb = 0; eb < 2; ++eb) {
      int et = wid + 8 * eb;                       // e-tile 0..15
      const _Float16* brow = Wp + (size_t)(16 * et + ln) * DM;
      v16h Bf[8];
#pragma unroll
      for (int kk = 0; kk < 8; ++kk) Bf[kk] = glb_b_frag(brow + 32 * kk, hi);
      v8f acc = {};
#pragma unroll
      for (int kk = 0; kk < 8; ++kk) acc = WMMA_F16(Af[kk], Bf[kk], acc);
      int col = 16 * et + ln;
      float bias = b_proj[col];
      float* orow = out + base;
#pragma unroll
      for (int r = 0; r < 8; ++r)
        orow[(16 * mt + r + 8 * hi) * DM + col] = acc[r] + bias;
    }
  }
}

extern "C" void kernel_launch(void* const* d_in, const int* in_sizes, int n_in,
                              void* d_out, int out_size, void* d_ws, size_t ws_size,
                              hipStream_t stream) {
  const float* x      = (const float*)d_in[0];
  const float* w_qkv  = (const float*)d_in[1];
  const float* b_qkv  = (const float*)d_in[2];
  const float* w_proj = (const float*)d_in[3];
  const float* b_proj = (const float*)d_in[4];
  float* out = (float*)d_out;

  _Float16* Wq = (_Float16*)d_ws;           // 768*256 f16
  _Float16* Wp = Wq + 3 * DM * DM;          // 256*256 f16

  // weights fp32 -> fp16 (L2-resident for the whole fused pass)
  wconvert_kernel<<<(3 * DM * DM + 255) / 256, 256, 0, stream>>>(w_qkv, w_proj, Wq, Wp);

  // fused per-(b,n) attention: 8*325 = 2600 workgroups, 8 waves each
  size_t lds_halves = (size_t)3 * SQ * PITCH + (size_t)NH * HD * VPITCH +
                      (size_t)NH * SQ * PPITCH;
  size_t lds_bytes = lds_halves * sizeof(_Float16);  // 207,872 B (<320KB/WGP)
  temporal_attn_kernel<<<8 * 325, 256, lds_bytes, stream>>>(x, b_qkv, b_proj, Wq, Wp, out);
}
